// Encoder_65317862637856
// MI455X (gfx1250) — matrix operand order repaired and checked by reading, not compiled, not run on hardware
//
#include <hip/hip_runtime.h>
#include <hip/hip_bf16.h>
#include <math.h>

// ---------------------------------------------------------------------------
// Pyraformer-style encoder forward for MI455X (gfx1250, wave32, WMMA).
// Heavy GEMMs: v_wmma_f32_16x16x32_bf16, weights pre-converted to bf16 once,
// per-chunk LDS staging via global_load_async_to_lds_b128 (ASYNCcnt),
// 64x64 macro-tiles, A-fragment reuse across 4 N-tiles.
// ---------------------------------------------------------------------------

typedef float  f32x8   __attribute__((ext_vector_type(8)));
typedef __bf16 bf16x16 __attribute__((ext_vector_type(16)));

#define BATCH  16
#define SEQL   512
#define DMODEL 512
#define DB     128
#define SKIP   680          // valid pyramid length 512+128+32+8
#define SP     688          // padded to 43*16
#define SPK    704          // padded to 22*32 (K dim for PV)
#define NHEAD  8
#define DK     64
#define DFF    2048

static __device__ __forceinline__ f32x8 zero8() {
  f32x8 z;
#pragma unroll
  for (int r = 0; r < 8; ++r) z[r] = 0.0f;
  return z;
}

// A fragment (16-bit 16x32 layout): lane = c*16+m holds K = c*16 + j, contiguous.
static __device__ __forceinline__ bf16x16 frag_a(const float* __restrict__ base) {
  bf16x16 a;
#pragma unroll
  for (int j = 0; j < 16; ++j) a[j] = (__bf16)base[j];
  return a;
}

// B fragment (16-bit 32x16 layout): lane = c*16+n holds K = c*16 + j, contiguous.
static __device__ __forceinline__ bf16x16 frag_b(const float* __restrict__ base) {
  bf16x16 b;
#pragma unroll
  for (int j = 0; j < 16; ++j) b[j] = (__bf16)base[j];
  return b;
}

static __device__ __forceinline__ f32x8 wmma_bf16(bf16x16 a, bf16x16 b, f32x8 c) {
  return __builtin_amdgcn_wmma_f32_16x16x32_bf16(false, a, false, b, (short)0, c,
                                                 false, false);
}

// Async DMA: 16 bytes global -> LDS, tracked by ASYNCcnt (ISA 15.18.3 op 98).
static __device__ __forceinline__ void async_g2l_b128(unsigned lds_addr,
                                                      const __bf16* gsrc) {
  asm volatile("global_load_async_to_lds_b128 %0, %1, off"
               :: "v"(lds_addr), "v"(gsrc) : "memory");
}
static __device__ __forceinline__ void wait_async0() {
  asm volatile("s_wait_asynccnt 0x0" ::: "memory");
}

// ---------------------------------------------------------------------------
// GEMM: C[M,N] = act(A[M,K] @ W[N,K]^T + bias + resid), W already bf16.
// Block = 128 threads (4 waves), 64x64 macro-tile: wave w -> rows
// [row0+16w,+16), all 64 cols (4 accumulators, A fragment reused 4x).
// Per 32-deep K chunk the 64x32 bf16 W sub-tile (4 KB) is DMA'd into LDS with
// global_load_async_to_lds_b128 (2 x 16B per thread), fenced by
// s_wait_asynccnt + barrier.  Requires M%64==0, N%64==0, K%32==0.
// act: 0 none, 1 exact GELU.
// ---------------------------------------------------------------------------
__global__ __launch_bounds__(128) void gemm_wmma(
    const float* __restrict__ A, int lda,
    const __bf16* __restrict__ W,           // [N,K] row-major, bf16
    const float* __restrict__ bias,         // [N] or null
    const float* __restrict__ resid,        // [M,ldc] or null
    float* __restrict__ C, int ldc,
    int K, int act) {
  __shared__ __align__(128) __bf16 wtile[64 * 32];

  const int tid  = threadIdx.x;
  const int lane = tid & 31, w = tid >> 5;
  const int c = lane >> 4, ln = lane & 15;
  const long row0 = (long)blockIdx.y * 64;
  const int  col0 = blockIdx.x * 64;

  const float* arow = A + (row0 + w * 16 + ln) * (long)lda;

  // Fill coords: thread t copies 16 contiguous bf16 (32B) of the 64x32 tile.
  const int fcol = tid >> 1;              // 0..63
  const int fk   = (tid & 1) * 16;        // 0 or 16
  const __bf16* fsrc = W + (long)(col0 + fcol) * K + fk;
  const unsigned ldst = (unsigned)(size_t)wtile + (unsigned)tid * 32u;

  f32x8 acc[4];
#pragma unroll
  for (int nt = 0; nt < 4; ++nt) acc[nt] = zero8();

  for (int k0 = 0; k0 < K; k0 += 32) {
    __syncthreads();                      // previous chunk's readers done
    async_g2l_b128(ldst, fsrc + k0);      // DMA W sub-tile into LDS
    async_g2l_b128(ldst + 16u, fsrc + k0 + 8);
    if (k0 + 32 < K) __builtin_prefetch(arow + k0 + 32, 0, 0);
    wait_async0();
    __syncthreads();

    bf16x16 af = frag_a(arow + k0 + c * 16);      // one A frag, 4 WMMAs
#pragma unroll
    for (int nt = 0; nt < 4; ++nt) {
      const bf16x16 bf =
          *reinterpret_cast<const bf16x16*>(&wtile[(nt * 16 + ln) * 32 + c * 16]);
      acc[nt] = wmma_bf16(af, bf, acc[nt]);
    }
  }

#pragma unroll
  for (int nt = 0; nt < 4; ++nt) {
    const int col = col0 + nt * 16 + ln;
    const float bv = bias ? bias[col] : 0.0f;
#pragma unroll
    for (int r = 0; r < 8; ++r) {
      long m = row0 + w * 16 + r + c * 8;  // C layout: VGPR r -> M=r+(lane>=16)*8
      float v = acc[nt][r] + bv;
      if (resid) v += resid[m * (long)ldc + col];
      if (act == 1) v = 0.5f * v * (1.0f + erff(v * 0.70710678118654752f));
      C[m * (long)ldc + col] = v;
    }
  }
}

// ---------------------------------------------------------------------------
// Flash-style masked attention for one (batch, head, 16-query tile).
// QK^T (WMMA) -> scale/mask into LDS -> softmax (wave32 shuffles) -> PV (WMMA).
// ---------------------------------------------------------------------------
__global__ __launch_bounds__(32) void attn_wmma(
    const float* __restrict__ qb,   // [B*SP, 512]
    const float* __restrict__ kb,   // [B*SP, 512]
    const float* __restrict__ vT,   // [B, H, DK, SPK] zero-padded cols
    const unsigned char* __restrict__ mask,   // [680,680] bool
    float* __restrict__ out) {      // [B*SP, 512]
  __shared__ float probs[16 * SPK];

  const int lane = threadIdx.x;
  const int c = lane >> 4, ln = lane & 15;
  const int qt = blockIdx.x, h = blockIdx.y, b = blockIdx.z;

  // Q fragments for this 16-row tile (K = 64 -> two 32-chunks), reused 43 times.
  const float* qrow = qb + ((long)b * SP + qt * 16 + ln) * DMODEL + h * DK;
  bf16x16 aq0 = frag_a(qrow + c * 16);
  bf16x16 aq1 = frag_a(qrow + 32 + c * 16);

  for (int nt = 0; nt < SP / 16; ++nt) {
    const float* krow = kb + ((long)b * SP + nt * 16 + ln) * DMODEL + h * DK;
    bf16x16 bk0 = frag_b(krow + c * 16);
    bf16x16 bk1 = frag_b(krow + 32 + c * 16);
    f32x8 acc = zero8();
    acc = wmma_bf16(aq0, bk0, acc);
    acc = wmma_bf16(aq1, bk1, acc);

    const int sk = nt * 16 + ln;
#pragma unroll
    for (int r = 0; r < 8; ++r) {
      int m  = r + c * 8;
      int sq = qt * 16 + m;
      float v = acc[r] * 0.125f;                    // 1/sqrt(DK)
      if (sq >= SKIP || sk >= SKIP) v = -1e30f;     // structural padding
      else if (mask[sq * SKIP + sk]) v = -1e9f;     // PAM mask (matches ref)
      probs[m * SPK + sk] = v;
    }
  }
  // K-padding columns 688..703 -> -1e30 (softmax to exactly zero weight)
#pragma unroll
  for (int r = 0; r < 8; ++r) probs[(r + c * 8) * SPK + SP + ln] = -1e30f;
  __syncthreads();

  // Softmax: lane pair {ln, ln+16} owns row ln; stride-2 columns each.
  {
    const int row = ln;
    float pm = -3.4e38f;
    for (int cc = c; cc < SPK; cc += 2) pm = fmaxf(pm, probs[row * SPK + cc]);
    pm = fmaxf(pm, __shfl_xor(pm, 16));
    float ps = 0.0f;
    for (int cc = c; cc < SPK; cc += 2) {
      float e = expf(probs[row * SPK + cc] - pm);
      probs[row * SPK + cc] = e;
      ps += e;
    }
    ps += __shfl_xor(ps, 16);
    float inv = 1.0f / ps;
    for (int cc = c; cc < SPK; cc += 2) probs[row * SPK + cc] *= inv;
  }
  __syncthreads();

  // PV: M=16 queries, N=64 (4 tiles of 16), K=704 from LDS probs vs vT.
  const float* vbase = vT + ((long)(b * NHEAD + h) * DK) * SPK;
  for (int nt2 = 0; nt2 < DK / 16; ++nt2) {
    const float* vrow = vbase + (long)(nt2 * 16 + ln) * SPK;
    f32x8 acc = zero8();
    for (int k0 = 0; k0 < SPK; k0 += 32) {
      bf16x16 ap = frag_a(&probs[ln * SPK + k0 + c * 16]);
      bf16x16 bv = frag_b(vrow + k0 + c * 16);
      acc = wmma_bf16(ap, bv, acc);
    }
#pragma unroll
    for (int r = 0; r < 8; ++r) {
      int m = r + c * 8;
      out[((long)b * SP + qt * 16 + m) * DMODEL + h * DK + nt2 * 16 + ln] = acc[r];
    }
  }
}

// ---------------------------------------------------------------------------
// Head: gather 4 pyramid scales per position (indexes) fused into A fragments,
// GEMM vs bf16 hd_w [512, 2048].  Same async-staged 64x64 macro-tile scheme.
// grid = (512/64, 8192/64), block = 128.
// ---------------------------------------------------------------------------
__global__ __launch_bounds__(128) void gemm_head(
    const float* __restrict__ x,            // [B*SP, 512]
    const int* __restrict__ indexes,        // [512*4]
    const __bf16* __restrict__ W,           // [512, 2048] bf16
    const float* __restrict__ bias,         // [512]
    float* __restrict__ out) {              // [B*512, 512]
  __shared__ __align__(128) __bf16 wtile[64 * 32];

  const int tid  = threadIdx.x;
  const int lane = tid & 31, w = tid >> 5;
  const int c = lane >> 4, ln = lane & 15;
  const long row0 = (long)blockIdx.y * 64;
  const int  col0 = blockIdx.x * 64;
  const int  K = 4 * DMODEL;

  const long lrow = row0 + w * 16 + ln;     // this lane's A row
  const int b = (int)(lrow >> 9);
  const int l = (int)(lrow & 511);

  const int fcol = tid >> 1;
  const int fk   = (tid & 1) * 16;
  const __bf16* fsrc = W + (long)(col0 + fcol) * K + fk;
  const unsigned ldst = (unsigned)(size_t)wtile + (unsigned)tid * 32u;

  f32x8 acc[4];
#pragma unroll
  for (int nt = 0; nt < 4; ++nt) acc[nt] = zero8();

  for (int k0 = 0; k0 < K; k0 += 32) {
    __syncthreads();
    async_g2l_b128(ldst, fsrc + k0);
    async_g2l_b128(ldst + 16u, fsrc + k0 + 8);
    wait_async0();
    __syncthreads();

    const int j = k0 >> 9;                      // scale index (chunk never crosses)
    const int srow = indexes[l * 4 + j];        // gather row in [0,680)
    const float* abase = x + ((long)(b * SP + srow)) * DMODEL + (k0 & 511);
    bf16x16 af = frag_a(abase + c * 16);
#pragma unroll
    for (int nt = 0; nt < 4; ++nt) {
      const bf16x16 bf =
          *reinterpret_cast<const bf16x16*>(&wtile[(nt * 16 + ln) * 32 + c * 16]);
      acc[nt] = wmma_bf16(af, bf, acc[nt]);
    }
  }

#pragma unroll
  for (int nt = 0; nt < 4; ++nt) {
    const int col = col0 + nt * 16 + ln;
    const float bv = bias[col];
#pragma unroll
    for (int r = 0; r < 8; ++r) {
      long m = row0 + w * 16 + r + c * 8;
      out[m * DMODEL + col] = acc[nt][r] + bv;
    }
  }
}

// ---------------------------------------------------------------------------
// Small VALU kernels
// ---------------------------------------------------------------------------

// One-time fp32 -> bf16 weight conversion.
__global__ void to_bf16(const float* __restrict__ src, __bf16* __restrict__ dst,
                        long n) {
  long i = (long)blockIdx.x * blockDim.x + threadIdx.x;
  if (i < n) dst[i] = (__bf16)src[i];
}

// Circular conv1d(k=3) token embedding + sinusoidal PE.
__global__ void embed_pe(const float* __restrict__ x_enc,   // [B,512,8]
                         const float* __restrict__ emb_w,   // [512,8,3]
                         float* __restrict__ seq) {         // [B,512,512]
  int idx = blockIdx.x * blockDim.x + threadIdx.x;
  if (idx >= BATCH * SEQL * DMODEL) return;
  int d = idx & 511, t = (idx >> 9) & 511, b = idx >> 18;
  float acc = 0.0f;
#pragma unroll
  for (int k = 0; k < 3; ++k) {
    int tt = (t + k - 1 + SEQL) & 511;          // circular pad
    const float* xp = x_enc + ((long)b * SEQL + tt) * 8;
#pragma unroll
    for (int ci = 0; ci < 8; ++ci) acc += xp[ci] * emb_w[(d * 8 + ci) * 3 + k];
  }
  float div = expf(-(float)(d & ~1) * (9.210340371976184f / 512.0f));
  float ang = (float)t * div;
  acc += (d & 1) ? cosf(ang) : sinf(ang);
  seq[idx] = acc;
}

// Strided conv (k=4, s=4) + eval-BatchNorm + ELU, layout [B, L, 128].
__global__ void conv_bn_elu(const float* __restrict__ in, int Lin,
                            float* __restrict__ outp, int Lout,
                            const float* __restrict__ w,    // [128,128,4]
                            const float* __restrict__ cb,
                            const float* __restrict__ g,
                            const float* __restrict__ bb) {
  int idx = blockIdx.x * blockDim.x + threadIdx.x;
  if (idx >= BATCH * Lout * DB) return;
  int o = idx & 127, t = (idx >> 7) % Lout, b = idx / (DB * Lout);
  float acc = 0.0f;
#pragma unroll
  for (int k = 0; k < 4; ++k) {
    const float* ip = in + ((long)b * Lin + 4 * t + k) * DB;
    for (int ci = 0; ci < DB; ++ci) acc += ip[ci] * w[(o * DB + ci) * 4 + k];
  }
  const float bnscale = rsqrtf(1.0f + 1e-5f);
  float v = (acc + cb[o]) * bnscale * g[o] + bb[o];
  outp[idx] = v > 0.0f ? v : (expf(v) - 1.0f);    // ELU
}

// Repack pyramid stages [B,128,*]+[B,32,*]+[B,8,*] -> [B,168,128] time-concat.
__global__ void repack_pyr(const float* __restrict__ p0, const float* __restrict__ p1,
                           const float* __restrict__ p2, float* __restrict__ upin) {
  int idx = blockIdx.x * blockDim.x + threadIdx.x;
  if (idx >= BATCH * 168 * DB) return;
  int ch = idx & 127, t = (idx >> 7) % 168, b = idx / (168 * DB);
  const float* src;
  if (t < 128)      src = p0 + ((long)b * 128 + t) * DB;
  else if (t < 160) src = p1 + ((long)b * 32 + (t - 128)) * DB;
  else              src = p2 + ((long)b * 8 + (t - 160)) * DB;
  upin[idx] = src[ch];
}

// Concat [seq | up] along time + LayerNorm (eps 1e-5) into padded x.
__global__ __launch_bounds__(32) void concat_ln(
    const float* __restrict__ seq, const float* __restrict__ up,
    const float* __restrict__ g, const float* __restrict__ bvec,
    float* __restrict__ x) {
  const int r = blockIdx.x;             // 0..B*SP-1
  const int b = r / SP, s = r % SP;
  float* dst = x + (long)r * DMODEL;
  const int lane = threadIdx.x;
  if (s >= SKIP) {                      // padding rows -> zero
    for (int i = lane; i < DMODEL; i += 32) dst[i] = 0.0f;
    return;
  }
  const float* src = (s < SEQL) ? seq + ((long)b * SEQL + s) * DMODEL
                                : up + ((long)b * 168 + (s - SEQL)) * DMODEL;
  float sm = 0.0f;
  for (int i = lane; i < DMODEL; i += 32) sm += src[i];
  for (int o = 16; o > 0; o >>= 1) sm += __shfl_xor(sm, o);
  float mean = sm * (1.0f / DMODEL);
  float vs = 0.0f;
  for (int i = lane; i < DMODEL; i += 32) { float d = src[i] - mean; vs += d * d; }
  for (int o = 16; o > 0; o >>= 1) vs += __shfl_xor(vs, o);
  float rstd = rsqrtf(vs * (1.0f / DMODEL) + 1e-5f);
  for (int i = lane; i < DMODEL; i += 32)
    dst[i] = (src[i] - mean) * rstd * g[i] + bvec[i];
}

// Generic per-row LayerNorm over 512 features (one wave per row).
__global__ __launch_bounds__(32) void layernorm_rows(
    const float* __restrict__ in, float* __restrict__ out,
    const float* __restrict__ g, const float* __restrict__ bvec, float eps) {
  const long r = blockIdx.x;
  const float* src = in + r * DMODEL;
  float* dst = out + r * DMODEL;
  const int lane = threadIdx.x;
  float sm = 0.0f;
  for (int i = lane; i < DMODEL; i += 32) sm += src[i];
  for (int o = 16; o > 0; o >>= 1) sm += __shfl_xor(sm, o);
  float mean = sm * (1.0f / DMODEL);
  float vs = 0.0f;
  for (int i = lane; i < DMODEL; i += 32) { float d = src[i] - mean; vs += d * d; }
  for (int o = 16; o > 0; o >>= 1) vs += __shfl_xor(vs, o);
  float rstd = rsqrtf(vs * (1.0f / DMODEL) + eps);
  for (int i = lane; i < DMODEL; i += 32)
    dst[i] = (src[i] - mean) * rstd * g[i] + bvec[i];
}

// Transpose V [B*SP, 512] -> vT [B, H, DK, SPK] with zero-padded columns.
__global__ void transpose_v(const float* __restrict__ vb, float* __restrict__ vT) {
  int idx = blockIdx.x * blockDim.x + threadIdx.x;
  if (idx >= BATCH * NHEAD * DK * SPK) return;
  int s = idx % SPK;
  int d = (idx / SPK) % DK;
  int h = (idx / (SPK * DK)) % NHEAD;
  int b = idx / (SPK * DK * NHEAD);
  vT[idx] = (s < SP) ? vb[((long)b * SP + s) * DMODEL + h * DK + d] : 0.0f;
}

// ---------------------------------------------------------------------------
// Host launcher
// ---------------------------------------------------------------------------
extern "C" void kernel_launch(void* const* d_in, const int* in_sizes, int n_in,
                              void* d_out, int out_size, void* d_ws, size_t ws_size,
                              hipStream_t stream) {
  (void)in_sizes; (void)n_in; (void)out_size; (void)ws_size;

  const float* x_enc  = (const float*)d_in[0];
  const float* emb_w  = (const float*)d_in[1];
  const float* down_w = (const float*)d_in[2];
  const float* down_b = (const float*)d_in[3];
  const float* conv_w = (const float*)d_in[4];
  const float* conv_b = (const float*)d_in[5];
  const float* bn_g   = (const float*)d_in[6];
  const float* bn_b   = (const float*)d_in[7];
  const float* up_w   = (const float*)d_in[8];
  const float* up_b   = (const float*)d_in[9];
  const float* bln_g  = (const float*)d_in[10];
  const float* bln_b  = (const float*)d_in[11];
  const float* wq     = (const float*)d_in[12];
  const float* wk     = (const float*)d_in[13];
  const float* wv     = (const float*)d_in[14];
  const float* fc_w   = (const float*)d_in[15];
  const float* fc_b   = (const float*)d_in[16];
  const float* ln1_g  = (const float*)d_in[17];
  const float* ln1_b  = (const float*)d_in[18];
  const float* w1     = (const float*)d_in[19];
  const float* b1     = (const float*)d_in[20];
  const float* w2     = (const float*)d_in[21];
  const float* b2     = (const float*)d_in[22];
  const float* ln2_g  = (const float*)d_in[23];
  const float* ln2_b  = (const float*)d_in[24];
  const float* hd_w   = (const float*)d_in[25];
  const float* hd_b   = (const float*)d_in[26];
  const unsigned char* mask = (const unsigned char*)d_in[27];
  const int* indexes  = (const int*)d_in[28];

  float* ws = (float*)d_ws;
  size_t off = 0;
  auto alloc = [&](size_t n) { float* p = ws + off; off += n; return p; };
  float* seq   = alloc((size_t)BATCH * SEQL * DMODEL);       // 4.19M
  float* down  = alloc((size_t)BATCH * SEQL * DB);           // 1.05M
  float* p0    = alloc((size_t)BATCH * 128 * DB);
  float* p1    = alloc((size_t)BATCH * 32 * DB);
  float* p2    = alloc((size_t)BATCH * 8 * DB);
  float* upin  = alloc((size_t)BATCH * 168 * DB);
  float* up    = alloc((size_t)BATCH * 168 * DMODEL);
  float* x     = alloc((size_t)BATCH * SP * DMODEL);         // 5.64M
  float* qb    = alloc((size_t)BATCH * SP * DMODEL);
  float* kb    = alloc((size_t)BATCH * SP * DMODEL);
  float* vb    = alloc((size_t)BATCH * SP * DMODEL);
  float* vT    = alloc((size_t)BATCH * NHEAD * DK * SPK);
  float* attn  = alloc((size_t)BATCH * SP * DMODEL);
  float* tmp   = alloc((size_t)BATCH * SP * DMODEL);
  float* ff    = alloc((size_t)BATCH * SP * DFF);            // 22.5M

  // bf16 weight arena (one-time conversion each launch; deterministic).
  __bf16* wbase = (__bf16*)(ws + off);
  size_t boff = 0;
  auto balloc = [&](size_t n) { __bf16* p = wbase + boff; boff += n; return p; };
  __bf16* down_wb = balloc((size_t)DB * DMODEL);
  __bf16* up_wb   = balloc((size_t)DMODEL * DB);
  __bf16* wq_b    = balloc((size_t)4 * DMODEL * DMODEL);
  __bf16* wk_b    = balloc((size_t)4 * DMODEL * DMODEL);
  __bf16* wv_b    = balloc((size_t)4 * DMODEL * DMODEL);
  __bf16* fc_wb   = balloc((size_t)4 * DMODEL * DMODEL);
  __bf16* w1_b    = balloc((size_t)4 * DFF * DMODEL);
  __bf16* w2_b    = balloc((size_t)4 * DMODEL * DFF);
  __bf16* hd_wb   = balloc((size_t)DMODEL * 4 * DMODEL);

  auto cvt = [&](const float* s, __bf16* d, long n) {
    to_bf16<<<(unsigned)((n + 255) / 256), 256, 0, stream>>>(s, d, n);
  };
  cvt(down_w, down_wb, (long)DB * DMODEL);
  cvt(up_w,   up_wb,   (long)DMODEL * DB);
  cvt(wq,     wq_b,    (long)4 * DMODEL * DMODEL);
  cvt(wk,     wk_b,    (long)4 * DMODEL * DMODEL);
  cvt(wv,     wv_b,    (long)4 * DMODEL * DMODEL);
  cvt(fc_w,   fc_wb,   (long)4 * DMODEL * DMODEL);
  cvt(w1,     w1_b,    (long)4 * DFF * DMODEL);
  cvt(w2,     w2_b,    (long)4 * DMODEL * DFF);
  cvt(hd_w,   hd_wb,   (long)DMODEL * 4 * DMODEL);

  const int ROWS = BATCH * SP;          // 11008 = 64*172

  // 1) embedding + PE
  {
    int n = BATCH * SEQL * DMODEL;
    embed_pe<<<(n + 255) / 256, 256, 0, stream>>>(x_enc, emb_w, seq);
  }
  // 2) bottleneck: down-projection
  gemm_wmma<<<dim3(DB / 64, (BATCH * SEQL) / 64), 128, 0, stream>>>(
      seq, DMODEL, down_wb, down_b, nullptr, down, DB, DMODEL, 0);
  // 3) pyramid convs
  {
    int n0 = BATCH * 128 * DB, n1 = BATCH * 32 * DB, n2 = BATCH * 8 * DB;
    conv_bn_elu<<<(n0 + 255) / 256, 256, 0, stream>>>(down, 512, p0, 128,
        conv_w, conv_b, bn_g, bn_b);
    conv_bn_elu<<<(n1 + 255) / 256, 256, 0, stream>>>(p0, 128, p1, 32,
        conv_w + 1 * DB * DB * 4, conv_b + DB, bn_g + DB, bn_b + DB);
    conv_bn_elu<<<(n2 + 255) / 256, 256, 0, stream>>>(p1, 32, p2, 8,
        conv_w + 2 * DB * DB * 4, conv_b + 2 * DB, bn_g + 2 * DB, bn_b + 2 * DB);
    int nr = BATCH * 168 * DB;
    repack_pyr<<<(nr + 255) / 256, 256, 0, stream>>>(p0, p1, p2, upin);
  }
  // 4) up-projection + concat + LN
  gemm_wmma<<<dim3(DMODEL / 64, (BATCH * 168) / 64), 128, 0, stream>>>(
      upin, DB, up_wb, up_b, nullptr, up, DMODEL, DB, 0);
  concat_ln<<<ROWS, 32, 0, stream>>>(seq, up, bln_g, bln_b, x);

  // 5) encoder layers
  for (int i = 0; i < 4; ++i) {
    const size_t wstep = (size_t)DMODEL * DMODEL;
    gemm_wmma<<<dim3(DMODEL / 64, ROWS / 64), 128, 0, stream>>>(
        x, DMODEL, wq_b + i * wstep, nullptr, nullptr, qb, DMODEL, DMODEL, 0);
    gemm_wmma<<<dim3(DMODEL / 64, ROWS / 64), 128, 0, stream>>>(
        x, DMODEL, wk_b + i * wstep, nullptr, nullptr, kb, DMODEL, DMODEL, 0);
    gemm_wmma<<<dim3(DMODEL / 64, ROWS / 64), 128, 0, stream>>>(
        x, DMODEL, wv_b + i * wstep, nullptr, nullptr, vb, DMODEL, DMODEL, 0);
    {
      int n = BATCH * NHEAD * DK * SPK;
      transpose_v<<<(n + 255) / 256, 256, 0, stream>>>(vb, vT);
    }
    attn_wmma<<<dim3(SP / 16, NHEAD, BATCH), 32, 0, stream>>>(qb, kb, vT, mask, attn);
    // fc + residual, then LN1
    gemm_wmma<<<dim3(DMODEL / 64, ROWS / 64), 128, 0, stream>>>(
        attn, DMODEL, fc_wb + i * wstep, fc_b + i * DMODEL,
        x, tmp, DMODEL, DMODEL, 0);
    layernorm_rows<<<ROWS, 32, 0, stream>>>(tmp, x, ln1_g + i * DMODEL,
                                            ln1_b + i * DMODEL, 1e-6f);
    // FFN: GELU(x@w1^T+b1) @ w2^T + b2 + residual, then LN2
    gemm_wmma<<<dim3(DFF / 64, ROWS / 64), 128, 0, stream>>>(
        x, DMODEL, w1_b + (size_t)i * DFF * DMODEL, b1 + i * DFF,
        nullptr, ff, DFF, DMODEL, 1);
    gemm_wmma<<<dim3(DMODEL / 64, ROWS / 64), 128, 0, stream>>>(
        ff, DFF, w2_b + (size_t)i * DMODEL * DFF, b2 + i * DMODEL,
        x, tmp, DMODEL, DFF, 0);
    layernorm_rows<<<ROWS, 32, 0, stream>>>(tmp, x, ln2_g + i * DMODEL,
                                            ln2_b + i * DMODEL, 1e-6f);
  }

  // 6) gather pyramid features + head projection (fused WMMA GEMM)
  gemm_head<<<dim3(DMODEL / 64, (BATCH * SEQL) / 64), 128, 0, stream>>>(
      x, indexes, hd_wb, hd_b, (float*)d_out);
}
